// ParityGameGATNetwork_5171140625123
// MI455X (gfx1250) — matrix operand order repaired and checked by reading, not compiled, not run on hardware
//
#include <hip/hip_runtime.h>
#include <cstdint>

// ---------------------------------------------------------------------------
// Types for CDNA5 WMMA (wave32)
// ---------------------------------------------------------------------------
typedef __attribute__((ext_vector_type(16))) _Float16 v16h;
typedef __attribute__((ext_vector_type(8)))  float    v8f;

__device__ __forceinline__ float sigf(float x) { return 1.0f / (1.0f + __expf(-x)); }

// order-preserving float -> uint mapping for atomicMax-based segment max
__device__ __forceinline__ unsigned fmapu(float f) {
    unsigned u = __float_as_uint(f);
    return (u >> 31) ? ~u : (u | 0x80000000u);
}
__device__ __forceinline__ float funmapu(unsigned u) {
    return (u >> 31) ? __uint_as_float(u & 0x7fffffffu) : __uint_as_float(~u);
}

// ---------------------------------------------------------------------------
// Utility kernels
// ---------------------------------------------------------------------------
__global__ __launch_bounds__(256) void k_zero(float* p, long n) {
    long i = (long)blockIdx.x * 256 + threadIdx.x;
    if (i < n) p[i] = 0.0f;
}

__global__ __launch_bounds__(256) void k_vadd(const float* __restrict__ a,
                                              const float* __restrict__ b,
                                              float* __restrict__ o, int n) {
    int i = blockIdx.x * 256 + threadIdx.x;
    if (i < n) o[i] = a[i] + b[i];
}

// layer-0 GAT linear: [N,3] @ [3,128]
__global__ __launch_bounds__(256) void k_lin3(const float* __restrict__ x,
                                              const float* __restrict__ W,
                                              float* __restrict__ out, int N) {
    int i = blockIdx.x * 256 + threadIdx.x;
    if (i >= N * 128) return;
    int n = i >> 7, c = i & 127;
    out[i] = x[n * 3 + 0] * W[c] + x[n * 3 + 1] * W[128 + c] + x[n * 3 + 2] * W[256 + c];
}

// ---------------------------------------------------------------------------
// Generic f16-WMMA GEMM:  C[M,N] (=|+=) act( A[M,K] * B + bias )
// B row-major [K,N] (transB=0) or [N,K] (transB=1).
// Block tile 128x128, 8 waves (wave w -> rows w*16, 8 n-tiles of 16), K-step 32.
// ---------------------------------------------------------------------------
__global__ __launch_bounds__(256)
void k_gemm(const float* __restrict__ A, int lda,
            const float* __restrict__ B, int ldb, int transB,
            const float* __restrict__ bias,
            float* __restrict__ C, int ldc,
            int M, int N, int K, int relu, int accum)
{
    __shared__ _Float16 As[128 * 36];           // [row][k], pad 4 halves
    __shared__ _Float16 Bs[128 * 36];           // transposed: [n][k]
    const int tid = threadIdx.x;
    const int wv = tid >> 5, lane = tid & 31;
    const int hi = lane >> 4, l15 = lane & 15;
    const int m_blk = blockIdx.x * 128;
    const int n_blk = blockIdx.y * 128;
    v8f acc[8] = {};

    for (int kk = 0; kk < K; kk += 32) {
        for (int i = tid; i < 128 * 32; i += 256) {
            int r = i >> 5, k = i & 31;
            int gm = m_blk + r;
            float v = (gm < M) ? A[(long)gm * lda + kk + k] : 0.0f;
            As[r * 36 + k] = (_Float16)v;
        }
        for (int i = tid; i < 128 * 32; i += 256) {
            int n = i >> 5, k = i & 31;
            int gn = n_blk + n;
            float v = 0.0f;
            if (gn < N)
                v = transB ? B[(long)gn * ldb + kk + k] : B[(long)(kk + k) * ldb + gn];
            Bs[n * 36 + k] = (_Float16)v;
        }
        __syncthreads();

        // A fragment: lane holds row m=wv*16+l15; half j -> k=(j&7)+16*(j>>3)+8*hi
        union { v16h h; uint32_t u[8]; } Af;
        const int arow = (wv * 16 + l15) * 36;
        #pragma unroll
        for (int i = 0; i < 8; i++) {
            int k0 = 2 * i + ((i >= 4) ? 8 : 0) + 8 * hi;
            Af.u[i] = *reinterpret_cast<const uint32_t*>(&As[arow + k0]);
        }
        #pragma unroll
        for (int nt = 0; nt < 8; nt++) {
            // B fragment: lane holds col n=nt*16+l15; half j -> k=(j&7)+16*(j>>3)+8*hi
            union { v16h h; uint32_t u[8]; } Bf;
            const int brow = (nt * 16 + l15) * 36 + 8 * hi;
            #pragma unroll
            for (int i = 0; i < 8; i++)
                Bf.u[i] = *reinterpret_cast<const uint32_t*>(&Bs[brow + 2 * i + ((i >= 4) ? 8 : 0)]);
            acc[nt] = __builtin_amdgcn_wmma_f32_16x16x32_f16(
                false, Af.h, false, Bf.h, (short)0, acc[nt], false, false);
        }
        __syncthreads();
    }

    // D layout: element r -> m = r + 8*hi, n = l15
    #pragma unroll
    for (int nt = 0; nt < 8; nt++) {
        int gn = n_blk + nt * 16 + l15;
        if (gn >= N) continue;
        float bv = bias ? bias[gn] : 0.0f;
        #pragma unroll
        for (int r = 0; r < 8; r++) {
            int gm = m_blk + wv * 16 + 8 * hi + r;
            if (gm < M) {
                float v = acc[nt][r] + bv;
                if (accum) v += C[(long)gm * ldc + gn];
                if (relu) v = fmaxf(v, 0.0f);
                C[(long)gm * ldc + gn] = v;
            }
        }
    }
}

// ---------------------------------------------------------------------------
// GAT helpers
// ---------------------------------------------------------------------------
// per-node dot products hsrc = h.a_src, hdst = h.a_dst (warp per node)
__global__ __launch_bounds__(256)
void k_node_dots(const float* __restrict__ h, const float* __restrict__ asrc,
                 const float* __restrict__ adst, float* __restrict__ hs,
                 float* __restrict__ hd, int N) {
    int n = blockIdx.x * 8 + (threadIdx.x >> 5);
    int lane = threadIdx.x & 31;
    if (n >= N) return;
    float s = 0.f, d = 0.f;
    #pragma unroll
    for (int i = 0; i < 4; i++) {
        float v = h[(long)n * 128 + lane + 32 * i];
        s += v * asrc[lane + 32 * i];
        d += v * adst[lane + 32 * i];
    }
    for (int off = 16; off > 0; off >>= 1) { s += __shfl_xor(s, off, 32); d += __shfl_xor(d, off, 32); }
    if (lane == 0) { hs[n] = s; hd[n] = d; }
}

__global__ __launch_bounds__(256)
void k_edge_score(const float* __restrict__ hs, const float* __restrict__ hd,
                  const int* __restrict__ row, const int* __restrict__ col,
                  float* __restrict__ ebuf, unsigned* __restrict__ mu, int E) {
    int e = blockIdx.x * 256 + threadIdx.x;
    if (e >= E) return;
    float v = hs[col[e]] + hd[row[e]];
    v = (v > 0.f) ? v : 0.2f * v;                 // leaky_relu(0.2)
    ebuf[e] = v;
    atomicMax(&mu[row[e]], fmapu(v));
}

__global__ __launch_bounds__(256)
void k_edge_exp(float* __restrict__ ebuf, const unsigned* __restrict__ mu,
                float* __restrict__ denom, const int* __restrict__ row, int E) {
    int e = blockIdx.x * 256 + threadIdx.x;
    if (e >= E) return;
    int r = row[e];
    float w = __expf(ebuf[e] - funmapu(mu[r]));
    ebuf[e] = w;
    unsafeAtomicAdd(&denom[r], w);
}

// warp per edge: acc[row] += alpha * h[col]
__global__ __launch_bounds__(256)
void k_aggregate(const float* __restrict__ h, const float* __restrict__ wbuf,
                 const float* __restrict__ denom, const int* __restrict__ row,
                 const int* __restrict__ col, float* __restrict__ acc, int E) {
    int e = blockIdx.x * 8 + (threadIdx.x >> 5);
    if (e >= E) return;
    int lane = threadIdx.x & 31;
    int r = row[e], c = col[e];
    float alpha = wbuf[e] / (denom[r] + 1e-16f);
    #pragma unroll
    for (int i = 0; i < 4; i++) {
        int ch = lane + 32 * i;
        unsafeAtomicAdd(&acc[(long)r * 128 + ch], alpha * h[(long)c * 128 + ch]);
    }
}

__global__ __launch_bounds__(256)
void k_bias_relu(float* __restrict__ p, const float* __restrict__ b, long n) {
    long i = (long)blockIdx.x * 256 + threadIdx.x;
    if (i < n) p[i] = fmaxf(p[i] + b[i & 127], 0.0f);
}

// ---------------------------------------------------------------------------
// LSTM pointwise step (gates layout [n][4H] in i,f,g,o order)
// ---------------------------------------------------------------------------
__global__ __launch_bounds__(256)
void k_lstm_step(const float* __restrict__ gates, float* __restrict__ cst,
                 float* __restrict__ hout, int N, int H) {
    long idx = (long)blockIdx.x * 256 + threadIdx.x;
    if (idx >= (long)N * H) return;
    long n = idx / H, j = idx % H;
    long base = n * 4 * H;
    float gi = gates[base + j];
    float gf = gates[base + H + j];
    float gg = gates[base + 2 * H + j];
    float go = gates[base + 3 * H + j];
    float c = sigf(gf) * cst[idx] + sigf(gi) * tanhf(gg);
    float h = sigf(go) * tanhf(c);
    cst[idx] = c;
    hout[idx] = h;
}

// ---------------------------------------------------------------------------
// JK attention: score[l][n] = <[hf,hb], attW> + b; softmax over L=3; blend xs
// ---------------------------------------------------------------------------
__global__ __launch_bounds__(256)
void k_jk(const float* __restrict__ hf, const float* __restrict__ hb,
          const float* __restrict__ attW, const float* __restrict__ attb,
          const float* __restrict__ xs, float* __restrict__ hout, int N, int H) {
    int n = blockIdx.x * 8 + (threadIdx.x >> 5);
    int lane = threadIdx.x & 31;
    if (n >= N) return;
    long NH = (long)N * H;
    float sc[3];
    #pragma unroll
    for (int l = 0; l < 3; l++) {
        float s = 0.f;
        for (int i = lane; i < H; i += 32) s += hf[l * NH + (long)n * H + i] * attW[i];
        for (int i = lane; i < H; i += 32) s += hb[l * NH + (long)n * H + i] * attW[H + i];
        for (int off = 16; off > 0; off >>= 1) s += __shfl_xor(s, off, 32);
        sc[l] = s + attb[0];
    }
    float mx = fmaxf(sc[0], fmaxf(sc[1], sc[2]));
    float a0 = __expf(sc[0] - mx), a1 = __expf(sc[1] - mx), a2 = __expf(sc[2] - mx);
    float inv = 1.0f / (a0 + a1 + a2);
    a0 *= inv; a1 *= inv; a2 *= inv;
    long NC = (long)N * 128;
    for (int c = lane; c < 128; c += 32) {
        long o = (long)n * 128 + c;
        hout[o] = a0 * xs[o] + a1 * xs[NC + o] + a2 * xs[2 * NC + o];
    }
}

// node head: [N,128] @ nW2[128,2] + nb2, softmax over 2
__global__ __launch_bounds__(256)
void k_node_head(const float* __restrict__ nh, const float* __restrict__ W2,
                 const float* __restrict__ b2, float* __restrict__ out, int N) {
    int n = blockIdx.x * 256 + threadIdx.x;
    if (n >= N) return;
    float z0 = b2[0], z1 = b2[1];
    for (int c = 0; c < 128; c++) {
        float v = nh[(long)n * 128 + c];
        z0 += v * W2[c * 2];
        z1 += v * W2[c * 2 + 1];
    }
    float mx = fmaxf(z0, z1);
    float e0 = __expf(z0 - mx), e1 = __expf(z1 - mx);
    float inv = 1.0f / (e0 + e1);
    out[(long)n * 2] = e0 * inv;
    out[(long)n * 2 + 1] = e1 * inv;
}

// ---------------------------------------------------------------------------
// Fused edge classifier: gather concat(h[row],h[col]) [128 edges x 256] ->
// WMMA GEMM with eW1 [256,128] -> bias/ReLU -> 128->2 head -> softmax.
// ---------------------------------------------------------------------------
__global__ __launch_bounds__(256)
void k_edge_mlp(const float* __restrict__ h, const int* __restrict__ row,
                const int* __restrict__ col,
                const float* __restrict__ eW1, const float* __restrict__ eb1,
                const float* __restrict__ eW2, const float* __restrict__ eb2,
                float* __restrict__ out, int E)
{
    __shared__ _Float16 As[128 * 36];
    __shared__ _Float16 Bs[128 * 36];
    const int tid = threadIdx.x;
    const int wv = tid >> 5, lane = tid & 31;
    const int hi = lane >> 4, l15 = lane & 15;
    const int e_blk = blockIdx.x * 128;
    v8f acc[8] = {};

    for (int kk = 0; kk < 256; kk += 32) {
        for (int i = tid; i < 128 * 32; i += 256) {
            int r = i >> 5, k = i & 31;
            int e = e_blk + r; if (e >= E) e = E - 1;
            int gk = kk + k;
            int node = (gk < 128) ? row[e] : col[e];
            As[r * 36 + k] = (_Float16)h[(long)node * 128 + (gk & 127)];
        }
        for (int i = tid; i < 128 * 32; i += 256) {
            int n = i >> 5, k = i & 31;
            Bs[n * 36 + k] = (_Float16)eW1[(long)(kk + k) * 128 + n];
        }
        __syncthreads();

        union { v16h h; uint32_t u[8]; } Af;
        const int arow = (wv * 16 + l15) * 36;
        #pragma unroll
        for (int i = 0; i < 8; i++) {
            int k0 = 2 * i + ((i >= 4) ? 8 : 0) + 8 * hi;
            Af.u[i] = *reinterpret_cast<const uint32_t*>(&As[arow + k0]);
        }
        #pragma unroll
        for (int nt = 0; nt < 8; nt++) {
            union { v16h h; uint32_t u[8]; } Bf;
            const int brow = (nt * 16 + l15) * 36 + 8 * hi;
            #pragma unroll
            for (int i = 0; i < 8; i++)
                Bf.u[i] = *reinterpret_cast<const uint32_t*>(&Bs[brow + 2 * i + ((i >= 4) ? 8 : 0)]);
            acc[nt] = __builtin_amdgcn_wmma_f32_16x16x32_f16(
                false, Af.h, false, Bf.h, (short)0, acc[nt], false, false);
        }
        __syncthreads();
    }

    // in-register head: per lane holds hidden n = nt*16+l15, edge m = r+8*hi
    float b2_0 = eb2[0], b2_1 = eb2[1];
    float b1v[8], w20[8], w21[8];
    #pragma unroll
    for (int nt = 0; nt < 8; nt++) {
        int n = nt * 16 + l15;
        b1v[nt] = eb1[n]; w20[nt] = eW2[n * 2]; w21[nt] = eW2[n * 2 + 1];
    }
    #pragma unroll
    for (int r = 0; r < 8; r++) {
        float s0 = 0.f, s1 = 0.f;
        #pragma unroll
        for (int nt = 0; nt < 8; nt++) {
            float hv = fmaxf(acc[nt][r] + b1v[nt], 0.0f);
            s0 += hv * w20[nt];
            s1 += hv * w21[nt];
        }
        // reduce over 16 lanes of the same half (covers all 128 hidden units)
        for (int off = 1; off < 16; off <<= 1) { s0 += __shfl_xor(s0, off, 32); s1 += __shfl_xor(s1, off, 32); }
        if (l15 == 0) {
            int e = e_blk + wv * 16 + 8 * hi + r;
            if (e < E) {
                float z0 = s0 + b2_0, z1 = s1 + b2_1;
                float mx = fmaxf(z0, z1);
                float e0 = __expf(z0 - mx), e1 = __expf(z1 - mx);
                float inv = 1.0f / (e0 + e1);
                out[(long)e * 2] = e0 * inv;
                out[(long)e * 2 + 1] = e1 * inv;
            }
        }
    }
}

// ---------------------------------------------------------------------------
// Host orchestration
// ---------------------------------------------------------------------------
extern "C" void kernel_launch(void* const* d_in, const int* in_sizes, int n_in,
                              void* d_out, int out_size, void* d_ws, size_t ws_size,
                              hipStream_t stream) {
    const int C = 128, H = 192, L = 3;
    const int N = in_sizes[0] / 3;
    const int E = in_sizes[1] / 2;

    const float* x       = (const float*)d_in[0];
    const int*   ei      = (const int*)d_in[1];
    const int*   row     = ei;
    const int*   col     = ei + E;
    const float* gat_W0  = (const float*)d_in[2];
    const float* gat_b0  = (const float*)d_in[3];
    const float* gat_as0 = (const float*)d_in[4];
    const float* gat_ad0 = (const float*)d_in[5];
    const float* gat_W   = (const float*)d_in[6];
    const float* gat_b   = (const float*)d_in[7];
    const float* gat_as  = (const float*)d_in[8];
    const float* gat_ad  = (const float*)d_in[9];
    const float* Wih_f   = (const float*)d_in[10];
    const float* Whh_f   = (const float*)d_in[11];
    const float* bih_f   = (const float*)d_in[12];
    const float* bhh_f   = (const float*)d_in[13];
    const float* Wih_b   = (const float*)d_in[14];
    const float* Whh_b   = (const float*)d_in[15];
    const float* bih_b   = (const float*)d_in[16];
    const float* bhh_b   = (const float*)d_in[17];
    const float* attW    = (const float*)d_in[18];
    const float* attb    = (const float*)d_in[19];
    const float* nW1     = (const float*)d_in[20];
    const float* nb1     = (const float*)d_in[21];
    const float* nW2     = (const float*)d_in[22];
    const float* nb2     = (const float*)d_in[23];
    const float* eW1     = (const float*)d_in[24];
    const float* eb1     = (const float*)d_in[25];
    const float* eW2     = (const float*)d_in[26];
    const float* eb2     = (const float*)d_in[27];

    float* out = (float*)d_out;

    // ---- workspace carve (floats) ----
    float* W = (float*)d_ws;
    long o = 0;
    const long NC = (long)N * C, NH = (long)N * H;
    float* hs     = W + o; o += 3 * NC;      // xs[0..2] after each GAT layer
    float* hlin   = W + o; o += NC;          // pre-aggregation h = x@W
    float* hsrc   = W + o; o += N;
    float* hdst   = W + o; o += N;
    float* denom  = W + o; o += N;
    unsigned* mu  = (unsigned*)(W + o); o += N;
    float* ebuf   = W + o; o += E;
    float* bias_f = W + o; o += 4 * H;
    float* bias_b = W + o; o += 4 * H;
    float* gates  = W + o; o += (long)N * 4 * H;
    float* h0     = W + o; o += NH;          // zero hidden for step 0
    float* cst    = W + o; o += NH;
    float* hf     = W + o; o += 3 * NH;
    float* hb     = W + o; o += 3 * NH;
    // post-LSTM buffers alias the (now-dead) gates region
    float* hjk = gates;                      // NC
    float* nh  = gates + NC;                 // NC
    (void)ws_size; (void)n_in; (void)out_size;

    auto zero = [&](float* p, long n) {
        k_zero<<<dim3((unsigned)((n + 255) / 256)), dim3(256), 0, stream>>>(p, n);
    };

    const int gemm_mblk = (N + 127) / 128;

    // ================= GAT stack (3 layers) =================
    for (int l = 0; l < L; l++) {
        if (l == 0) {
            k_lin3<<<dim3((unsigned)((NC + 255) / 256)), dim3(256), 0, stream>>>(x, gat_W0, hlin, N);
        } else {
            k_gemm<<<dim3(gemm_mblk, 1), dim3(256), 0, stream>>>(
                hs + (long)(l - 1) * NC, C, gat_W + (long)(l - 1) * C * C, C, 0,
                nullptr, hlin, C, N, C, C, 0, 0);
        }
        const float* as_l = (l == 0) ? gat_as0 : gat_as + (long)(l - 1) * C;
        const float* ad_l = (l == 0) ? gat_ad0 : gat_ad + (long)(l - 1) * C;
        const float* b_l  = (l == 0) ? gat_b0  : gat_b  + (long)(l - 1) * C;

        k_node_dots<<<dim3((N + 7) / 8), dim3(256), 0, stream>>>(hlin, as_l, ad_l, hsrc, hdst, N);
        zero((float*)mu, N);
        zero(denom, N);
        zero(hs + (long)l * NC, NC);
        k_edge_score<<<dim3((E + 255) / 256), dim3(256), 0, stream>>>(hsrc, hdst, row, col, ebuf, mu, E);
        k_edge_exp<<<dim3((E + 255) / 256), dim3(256), 0, stream>>>(ebuf, mu, denom, row, E);
        k_aggregate<<<dim3((E + 7) / 8), dim3(256), 0, stream>>>(hlin, ebuf, denom, row, col,
                                                                 hs + (long)l * NC, E);
        k_bias_relu<<<dim3((unsigned)((NC + 255) / 256)), dim3(256), 0, stream>>>(
            hs + (long)l * NC, b_l, NC);
    }

    // ================= JK bidirectional LSTM over layers =================
    k_vadd<<<dim3(3), dim3(256), 0, stream>>>(bih_f, bhh_f, bias_f, 4 * H);
    k_vadd<<<dim3(3), dim3(256), 0, stream>>>(bih_b, bhh_b, bias_b, 4 * H);
    zero(h0, NH);
    const long NGH = (long)N * H;

    // forward direction: l = 0,1,2
    zero(cst, NH);
    for (int s = 0; s < 3; s++) {
        k_gemm<<<dim3(gemm_mblk, 6), dim3(256), 0, stream>>>(
            hs + (long)s * NC, C, Wih_f, C, 1, bias_f, gates, 4 * H, N, 4 * H, C, 0, 0);
        const float* hin = (s == 0) ? h0 : hf + (long)(s - 1) * NGH;
        k_gemm<<<dim3(gemm_mblk, 6), dim3(256), 0, stream>>>(
            hin, H, Whh_f, H, 1, nullptr, gates, 4 * H, N, 4 * H, H, 0, 1);
        k_lstm_step<<<dim3((unsigned)((NGH + 255) / 256)), dim3(256), 0, stream>>>(
            gates, cst, hf + (long)s * NGH, N, H);
    }
    // backward direction: l = 2,1,0
    zero(cst, NH);
    for (int s = 0; s < 3; s++) {
        int l = 2 - s;
        k_gemm<<<dim3(gemm_mblk, 6), dim3(256), 0, stream>>>(
            hs + (long)l * NC, C, Wih_b, C, 1, bias_b, gates, 4 * H, N, 4 * H, C, 0, 0);
        const float* hin = (s == 0) ? h0 : hb + (long)(l + 1) * NGH;
        k_gemm<<<dim3(gemm_mblk, 6), dim3(256), 0, stream>>>(
            hin, H, Whh_b, H, 1, nullptr, gates, 4 * H, N, 4 * H, H, 0, 1);
        k_lstm_step<<<dim3((unsigned)((NGH + 255) / 256)), dim3(256), 0, stream>>>(
            gates, cst, hb + (long)l * NGH, N, H);
    }

    // attention over layers -> hjk [N,C]
    k_jk<<<dim3((N + 7) / 8), dim3(256), 0, stream>>>(hf, hb, attW, attb, hs, hjk, N, H);

    // ================= node classifier =================
    k_gemm<<<dim3(gemm_mblk, 1), dim3(256), 0, stream>>>(
        hjk, C, nW1, C, 0, nb1, nh, C, N, C, C, 1, 0);
    k_node_head<<<dim3((N + 255) / 256), dim3(256), 0, stream>>>(nh, nW2, nb2, out, N);

    // ================= fused edge classifier =================
    k_edge_mlp<<<dim3((E + 127) / 128), dim3(256), 0, stream>>>(
        hjk, row, col, eW1, eb1, eW2, eb2, out + (long)N * 2, E);
}
